// ModelVAE_42417097016194
// MI455X (gfx1250) — matrix operand order repaired and checked by reading, not compiled, not run on hardware
//
#include <hip/hip_runtime.h>

// ---------------------------------------------------------------------------
// MI455X (gfx1250) VAE forward.
// ~247 GFLOP of GEMM vs ~400MB traffic -> compute bound on the matrix units
// -> all GEMMs via v_wmma_f32_16x16x32_f16 (f32 accumulate), activations f16.
// Weights pre-transposed to column-major f16 so both tile stages are straight
// 16B-per-lane copies, staged via CDNA5 GLOBAL_LOAD_ASYNC_TO_LDS_B128
// (ASYNCcnt) -- builtin confirmed present; args are v4i AS1* / v4i AS3*.
// ---------------------------------------------------------------------------

typedef __attribute__((ext_vector_type(16))) _Float16 v16h;
typedef __attribute__((ext_vector_type(8)))  _Float16 v8h;
typedef __attribute__((ext_vector_type(8)))  float    v8f;
typedef __attribute__((ext_vector_type(4)))  int      v4i;

#define BM 128
#define BN 128
#define BK 32
#define LDST 40   // LDS row stride in halves (padded to dodge bank conflicts)

#if __has_builtin(__builtin_amdgcn_global_load_async_to_lds_b128) && \
    __has_builtin(__builtin_amdgcn_s_wait_asynccnt)
#define USE_ASYNC_LDS 1
#else
#define USE_ASYNC_LDS 0
#endif

#define AS1 __attribute__((address_space(1)))
#define AS3 __attribute__((address_space(3)))

// Addrspace casts routed through integers (clang C++ dislikes direct AS casts).
// Global: flat VA == AS1 VA. LDS: low 32 bits of a flat LDS pointer are the
// AS3 offset (ISA 10.2).
__device__ __forceinline__ AS1 v4i* to_global_v4i(const void* p) {
    return (AS1 v4i*)(unsigned long long)(__SIZE_TYPE__)p;
}
__device__ __forceinline__ AS3 v4i* to_lds_v4i(const void* p) {
    return (AS3 v4i*)(unsigned int)(__SIZE_TYPE__)p;
}

// ---------------------------- wave helpers ---------------------------------
__device__ __forceinline__ float wsum(float v) {
    #pragma unroll
    for (int o = 16; o > 0; o >>= 1) v += __shfl_xor(v, o, 32);
    return v;
}
__device__ __forceinline__ int wmin(int v) {
    #pragma unroll
    for (int o = 16; o > 0; o >>= 1) v = min(v, __shfl_xor(v, o, 32));
    return v;
}
__device__ __forceinline__ unsigned hash32(unsigned x) {
    x ^= x >> 16; x *= 0x7feb352du;
    x ^= x >> 15; x *= 0x846ca68bu;
    x ^= x >> 16;
    return x;
}
__device__ __forceinline__ float urand(unsigned row, unsigned k, unsigned salt) {
    unsigned h = hash32(hash32(row * 0x9E3779B9u ^ salt) + k * 0x85EBCA6Bu);
    return ((h >> 8) + 0.5f) * (1.0f / 16777216.0f);
}

// ------------------------- fp32 -> f16 convert/pad -------------------------
__global__ void cvt_pad(const float* __restrict__ in, _Float16* __restrict__ out,
                        int R, int C, int RP, int CP) {
    size_t i = (size_t)blockIdx.x * blockDim.x + threadIdx.x;
    size_t tot = (size_t)RP * CP;
    if (i >= tot) return;
    int r = (int)(i / CP), c = (int)(i % CP);
    out[i] = (r < R && c < C) ? (_Float16)in[(size_t)r * C + c] : (_Float16)0.f;
}

// Transpose + pad: in [R=K rows, C=N cols] row-major fp32 -> Bt[NP][KP] f16.
__global__ void cvt_pad_t(const float* __restrict__ in, _Float16* __restrict__ out,
                          int R, int C, int KP, int NP) {
    size_t i = (size_t)blockIdx.x * blockDim.x + threadIdx.x;
    size_t tot = (size_t)NP * KP;
    if (i >= tot) return;
    int n = (int)(i / KP), k = (int)(i % KP);
    out[i] = (k < R && n < C) ? (_Float16)in[(size_t)k * C + n] : (_Float16)0.f;
}

// Pack W_mean [1024x64] + W_var [1024x1] into transposed f16 [128][1024].
__global__ void pack_mean_var_t(const float* __restrict__ Wm, const float* __restrict__ Wv,
                                _Float16* __restrict__ out) {
    int i = blockIdx.x * blockDim.x + threadIdx.x;
    if (i >= 128 * 1024) return;
    int n = i >> 10, k = i & 1023;
    float v = 0.f;
    if (n < 64)       v = Wm[k * 64 + n];
    else if (n == 64) v = Wv[k];
    out[i] = (_Float16)v;
}
__global__ void pack_bias_mv(const float* __restrict__ bm, const float* __restrict__ bv,
                             float* __restrict__ out) {
    int c = threadIdx.x;
    if (c >= 80) return;
    out[c] = (c < 64) ? bm[c] : ((c == 64) ? bv[0] : 0.f);
}

// ------------------------------- WMMA GEMM ---------------------------------
// C[M,realN] = act(A[M,Kpad] * B + bias); A row-major f16, Bt column-major
// f16 [Npad][Kpad]. 256 threads = 8 waves; block tile 128x128; wave tile
// 64x32 = 8 v_wmma_f32_16x16x32_f16 accumulators.
template<bool RELU>
__global__ __launch_bounds__(256)
void gemm_wmma(const _Float16* __restrict__ A, const _Float16* __restrict__ Bt,
               const float* __restrict__ bias,
               float* __restrict__ outF, _Float16* __restrict__ outH,
               int M, int Kpad, int realN, int ldcF, int ldcH)
{
    __shared__ __align__(16) _Float16 As[2][BM][LDST];  // [m][k]
    __shared__ __align__(16) _Float16 Bs[2][BN][LDST];  // [n][k]

    const int tid  = threadIdx.x;
    const int lane = tid & 31;
    const int wave = tid >> 5;
    const int row0 = blockIdx.y * BM;
    const int col0 = blockIdx.x * BN;

    // staging indices: each thread copies 32B of A and 32B of B per K-step
    const int sr = tid >> 1;           // tile row (m or n) 0..127
    const int sc = (tid & 1) * 16;     // k offset 0 / 16

    // fragment lane mapping: m/n = lane%16, k = 16*(lane/16)+s for A and B alike
    const int mW = (wave >> 2) * 64;
    const int nW = (wave & 3) * 32;
    const int mL = lane & 15;
    const int kA = (lane < 16) ? 0 : 16;   // A: K base per half-wave
    const int kB = (lane < 16) ? 0 : 16;   // B: K base per half-wave

    v8f acc[4][2];
    const v8f vzero = {0.f,0.f,0.f,0.f,0.f,0.f,0.f,0.f};
    #pragma unroll
    for (int mt = 0; mt < 4; mt++)
        #pragma unroll
        for (int nt = 0; nt < 2; nt++) acc[mt][nt] = vzero;

    const _Float16* gA = A  + (size_t)(row0 + sr) * Kpad + sc;
    const _Float16* gB = Bt + (size_t)(col0 + sr) * Kpad + sc;

#if USE_ASYNC_LDS
    // CDNA5 async copy: cache -> LDS without touching VGPRs (ASYNCcnt).
    auto stage = [&](int k0, int buf) {
        __builtin_amdgcn_global_load_async_to_lds_b128(
            to_global_v4i(gA + k0),     to_lds_v4i(&As[buf][sr][sc]),     0, 0);
        __builtin_amdgcn_global_load_async_to_lds_b128(
            to_global_v4i(gA + k0 + 8), to_lds_v4i(&As[buf][sr][sc + 8]), 0, 0);
        __builtin_amdgcn_global_load_async_to_lds_b128(
            to_global_v4i(gB + k0),     to_lds_v4i(&Bs[buf][sr][sc]),     0, 0);
        __builtin_amdgcn_global_load_async_to_lds_b128(
            to_global_v4i(gB + k0 + 8), to_lds_v4i(&Bs[buf][sr][sc + 8]), 0, 0);
    };
#else
    v8h rA0, rA1, rB0, rB1;
    auto loadTile = [&](int k0) {
        rA0 = *(const v8h*)(gA + k0);
        rA1 = *(const v8h*)(gA + k0 + 8);
        rB0 = *(const v8h*)(gB + k0);
        rB1 = *(const v8h*)(gB + k0 + 8);
    };
    auto stageTile = [&](int buf) {
        *(v8h*)&As[buf][sr][sc]     = rA0;
        *(v8h*)&As[buf][sr][sc + 8] = rA1;
        *(v8h*)&Bs[buf][sr][sc]     = rB0;
        *(v8h*)&Bs[buf][sr][sc + 8] = rB1;
    };
#endif

    auto compute = [&](int buf) {
        v16h af[4], bf[2];
        #pragma unroll
        for (int mt = 0; mt < 4; mt++) {
            const _Float16* p = &As[buf][mW + mt * 16 + mL][kA];
            v8h lo = *(const v8h*)p;          // K = kA .. kA+7
            v8h hi = *(const v8h*)(p + 8);    // K = kA+8 .. kA+15
            af[mt] = __builtin_shufflevector(lo, hi, 0,1,2,3,4,5,6,7,8,9,10,11,12,13,14,15);
        }
        #pragma unroll
        for (int nt = 0; nt < 2; nt++) {
            const _Float16* p = &Bs[buf][nW + nt * 16 + mL][kB];
            v8h lo = *(const v8h*)p;          // K = kB .. kB+7
            v8h hi = *(const v8h*)(p + 8);    // K = kB+8 .. kB+15
            bf[nt] = __builtin_shufflevector(lo, hi, 0,1,2,3,4,5,6,7,8,9,10,11,12,13,14,15);
        }
        #pragma unroll
        for (int mt = 0; mt < 4; mt++)
            #pragma unroll
            for (int nt = 0; nt < 2; nt++)
                acc[mt][nt] = __builtin_amdgcn_wmma_f32_16x16x32_f16(
                    false, af[mt], false, bf[nt], (short)0, acc[mt][nt], false, false);
    };

    const int nk = Kpad / BK;
#if USE_ASYNC_LDS
    stage(0, 0);
    __builtin_amdgcn_s_wait_asynccnt(0);
    __syncthreads();
    for (int kt = 0; kt < nk; kt++) {
        const int cur = kt & 1;
        if (kt + 1 < nk) stage((kt + 1) * BK, cur ^ 1);   // overlap with compute
        compute(cur);
        if (kt + 1 < nk) __builtin_amdgcn_s_wait_asynccnt(0);
        __syncthreads();
    }
#else
    loadTile(0); stageTile(0);
    __syncthreads();
    for (int kt = 0; kt < nk; kt++) {
        const int cur = kt & 1;
        if (kt + 1 < nk) loadTile((kt + 1) * BK);         // overlap with compute
        compute(cur);
        if (kt + 1 < nk) stageTile(cur ^ 1);
        __syncthreads();
    }
#endif

    // epilogue: C/D layout -> lanes 0-15: M=v, N=lane; lanes 16-31: M=v+8
    const int mHi = (lane >> 4) * 8;
    #pragma unroll
    for (int mt = 0; mt < 4; mt++)
        #pragma unroll
        for (int nt = 0; nt < 2; nt++)
            #pragma unroll
            for (int v = 0; v < 8; v++) {
                int gr = row0 + mW + mt * 16 + v + mHi;
                int gc = col0 + nW + nt * 16 + mL;
                if (gc < realN) {
                    float val = acc[mt][nt][v] + bias[gc];
                    if (RELU) val = fmaxf(val, 0.f);
                    if (outF) outF[(size_t)gr * ldcF + gc] = val;
                    if (outH) outH[(size_t)gr * ldcH + gc] = (_Float16)val;
                }
            }
}

// ------------------- z_mean normalize + vMF rsample ------------------------
// One wave32 per row; lane owns components {2l, 2l+1} of the 64-dim vectors.
__global__ __launch_bounds__(256)
void vmf_post(const float* __restrict__ mv, float* __restrict__ o_zmean,
              float* __restrict__ o_zvar, float* __restrict__ o_z,
              _Float16* __restrict__ zh, int B)
{
    const int lane = threadIdx.x & 31;
    const int row  = blockIdx.x * 8 + (threadIdx.x >> 5);
    if (row >= B) return;
    const int c0 = 2 * lane, c1 = 2 * lane + 1;

    // normalize z_mean
    float m0 = mv[row * 80 + c0], m1 = mv[row * 80 + c1];
    float nrm = sqrtf(wsum(m0 * m0 + m1 * m1)) + 1e-12f;
    float zm0 = m0 / nrm, zm1 = m1 / nrm;
    o_zmean[row * 64 + c0] = zm0;
    o_zmean[row * 64 + c1] = zm1;

    // kappa = softplus(raw) + 1
    float vr = mv[row * 80 + 64];
    float kappa = ((vr > 20.f) ? vr : log1pf(expf(vr))) + 1.0f;
    if (lane == 0) o_zvar[row] = kappa;

    // envelope params (Ulrich/Davidson), m=64, fm1=63
    const float fm1 = 63.f;
    float cc     = sqrtf(4.f * kappa * kappa + fm1 * fm1);
    float b_true = (-2.f * kappa + cc) / fm1;
    float b_app  = fm1 / (4.f * kappa);
    float s      = fminf(fmaxf(kappa - 10.f, 0.f), 1.f);
    float bb     = b_app * s + b_true * (1.f - s);
    float aa     = (fm1 + 2.f * kappa + cc) * 0.25f;
    float dd     = 4.f * aa * bb / (1.f + bb) - fm1 * logf(fm1);

    // K=32 proposals, one per lane; Beta(31.5,31.5) via normal approx
    float p1 = urand(row, lane * 4 + 0, 0x1111u);
    float p2 = urand(row, lane * 4 + 1, 0x1111u);
    float n0 = sqrtf(-2.f * logf(fmaxf(p1, 1e-20f))) * cosf(6.28318530718f * p2);
    float e  = fminf(fmaxf(0.5f + 0.0625f * n0, 1e-6f), 1.f - 1e-6f);
    float uu = fminf(fmaxf(urand(row, lane * 4 + 2, 0x1111u), 1e-20f), 1.f - 1e-7f);
    float den = 1.f - (1.f - bb) * e;
    float w   = (1.f - (1.f + bb) * e) / den;
    float t   = 2.f * aa * bb / den;
    bool  acc = (fm1 * logf(fmaxf(t, 1e-30f)) - t + dd) > logf(uu);
    int sel = wmin(acc ? lane : 32);
    if (sel == 32) sel = 0;
    w = __shfl(w, sel, 32);

    // tangent direction: 63-dim gaussian, normalized
    float g0 = 0.f, g1 = 0.f;
    {
        float a1 = urand(row, 128 + c0 * 2, 0x2222u), a2 = urand(row, 129 + c0 * 2, 0x2222u);
        float b1 = urand(row, 128 + c1 * 2, 0x2222u), b2 = urand(row, 129 + c1 * 2, 0x2222u);
        if (c0 < 63) g0 = sqrtf(-2.f * logf(fmaxf(a1, 1e-20f))) * cosf(6.28318530718f * a2);
        if (c1 < 63) g1 = sqrtf(-2.f * logf(fmaxf(b1, 1e-20f))) * cosf(6.28318530718f * b2);
    }
    float vn = rsqrtf(wsum(g0 * g0 + g1 * g1) + 1e-20f);
    g0 *= vn; g1 *= vn;
    float wp = sqrtf(fmaxf(1.f - w * w, 1e-10f));

    // xs[c] = c==0 ? w : wp * v[c-1]; v[2l-1] lives in lane l-1's g1
    float gprev = __shfl(g1, (lane + 31) & 31, 32);
    float xs0 = (lane == 0) ? w : wp * gprev;
    float xs1 = wp * g0;

    // Householder e1 -> loc
    float u0  = ((lane == 0) ? 1.f : 0.f) - zm0;
    float u1c = -zm1;
    float un  = sqrtf(wsum(u0 * u0 + u1c * u1c)) + 1e-5f;
    u0 /= un; u1c /= un;
    float dot = wsum(xs0 * u0 + xs1 * u1c);
    float z0 = xs0 - 2.f * dot * u0;
    float z1 = xs1 - 2.f * dot * u1c;

    o_z[row * 64 + c0] = z0;
    o_z[row * 64 + c1] = z1;
    zh[row * 64 + c0] = (_Float16)z0;
    zh[row * 64 + c1] = (_Float16)z1;
}

// ------------------------------ host driver --------------------------------
extern "C" void kernel_launch(void* const* d_in, const int* in_sizes, int n_in,
                              void* d_out, int out_size, void* d_ws, size_t ws_size,
                              hipStream_t stream) {
    (void)in_sizes; (void)n_in; (void)out_size; (void)ws_size;
    const int B = 16384, H = 1024, Z = 64, IN = 784;
    const int INP = 800;    // 784 padded to x32 (K of encoder GEMM 1)
    const int INNP = 896;   // 784 padded to x128 (N of logits GEMM)

    const float* x   = (const float*)d_in[0];
    const float* We0 = (const float*)d_in[1];  const float* be0 = (const float*)d_in[2];
    const float* We1 = (const float*)d_in[3];  const float* be1 = (const float*)d_in[4];
    const float* Wm  = (const float*)d_in[5];  const float* bm  = (const float*)d_in[6];
    const float* Wv  = (const float*)d_in[7];  const float* bv  = (const float*)d_in[8];
    const float* Wd0 = (const float*)d_in[9];  const float* bd0 = (const float*)d_in[10];
    const float* Wd1 = (const float*)d_in[11]; const float* bd1 = (const float*)d_in[12];
    const float* Wl  = (const float*)d_in[13]; const float* bl  = (const float*)d_in[14];

    // workspace carve-up (all 256B aligned)
    char* ws = (char*)d_ws;
    size_t off = 0;
    auto carve = [&](size_t bytes) -> char* {
        char* p = ws + off;
        off = (off + bytes + 255) & ~(size_t)255;
        return p;
    };
    _Float16* xh    = (_Float16*)carve((size_t)B * INP * 2);
    _Float16* we0t  = (_Float16*)carve((size_t)2 * H * INP * 2);     // [2048][800]
    _Float16* we1t  = (_Float16*)carve((size_t)H * 2 * H * 2);       // [1024][2048]
    _Float16* wmvt  = (_Float16*)carve((size_t)128 * H * 2);         // [128][1024]
    float*    bmv   = (float*)   carve(80 * 4);
    _Float16* wd0t  = (_Float16*)carve((size_t)H * Z * 2);           // [1024][64]
    _Float16* wd1t  = (_Float16*)carve((size_t)2 * H * H * 2);       // [2048][1024]
    _Float16* wlt   = (_Float16*)carve((size_t)INNP * 2 * H * 2);    // [896][2048]
    _Float16* h0h   = (_Float16*)carve((size_t)B * 2 * H * 2);
    _Float16* h1h   = (_Float16*)carve((size_t)B * H * 2);
    float*    mv    = (float*)   carve((size_t)B * 80 * 4);
    _Float16* zhbuf = (_Float16*)carve((size_t)B * Z * 2);
    _Float16* hd0h  = (_Float16*)carve((size_t)B * H * 2);
    _Float16* hd1h  = (_Float16*)carve((size_t)B * 2 * H * 2);

    float* o_zmean = (float*)d_out;
    float* o_zvar  = o_zmean + (size_t)B * Z;
    float* o_z     = o_zvar + B;
    float* o_xr    = o_z + (size_t)B * Z;

    auto cvtGrid = [](size_t tot) { return dim3((unsigned)((tot + 255) / 256)); };

    // fp32 -> f16: activations row-major padded; weights transposed+padded
    cvt_pad<<<cvtGrid((size_t)B * INP), 256, 0, stream>>>(x, xh, B, IN, B, INP);
    cvt_pad_t<<<cvtGrid((size_t)2 * H * INP), 256, 0, stream>>>(We0, we0t, IN, 2 * H, INP, 2 * H);
    cvt_pad_t<<<cvtGrid((size_t)H * 2 * H), 256, 0, stream>>>(We1, we1t, 2 * H, H, 2 * H, H);
    pack_mean_var_t<<<cvtGrid((size_t)128 * H), 256, 0, stream>>>(Wm, Wv, wmvt);
    pack_bias_mv<<<1, 128, 0, stream>>>(bm, bv, bmv);
    cvt_pad_t<<<cvtGrid((size_t)H * Z), 256, 0, stream>>>(Wd0, wd0t, Z, H, Z, H);
    cvt_pad_t<<<cvtGrid((size_t)2 * H * H), 256, 0, stream>>>(Wd1, wd1t, H, 2 * H, H, 2 * H);
    cvt_pad_t<<<cvtGrid((size_t)INNP * 2 * H), 256, 0, stream>>>(Wl, wlt, 2 * H, IN, 2 * H, INNP);

    // encoder
    gemm_wmma<true><<<dim3(2 * H / BN, B / BM), 256, 0, stream>>>(
        xh, we0t, be0, nullptr, h0h, B, INP, 2 * H, 0, 2 * H);
    gemm_wmma<true><<<dim3(H / BN, B / BM), 256, 0, stream>>>(
        h0h, we1t, be1, nullptr, h1h, B, 2 * H, H, 0, H);
    // fused z_mean (cols 0..63) + z_var raw (col 64)
    gemm_wmma<false><<<dim3(1, B / BM), 256, 0, stream>>>(
        h1h, wmvt, bmv, mv, nullptr, B, H, 65, 80, 0);

    // normalize + vMF reparameterized sample
    vmf_post<<<B / 8, 256, 0, stream>>>(mv, o_zmean, o_zvar, o_z, zhbuf, B);

    // decoder
    gemm_wmma<true><<<dim3(H / BN, B / BM), 256, 0, stream>>>(
        zhbuf, wd0t, bd0, nullptr, hd0h, B, Z, H, 0, H);
    gemm_wmma<true><<<dim3(2 * H / BN, B / BM), 256, 0, stream>>>(
        hd0h, wd1t, bd1, nullptr, hd1h, B, H, 2 * H, 0, 2 * H);
    gemm_wmma<false><<<dim3(INNP / BN, B / BM), 256, 0, stream>>>(
        hd1h, wlt, bl, o_xr, nullptr, B, 2 * H, IN, IN, 0);
}
